// StabilizedEventGroupExpertResidualConditioner_85907935855403
// MI455X (gfx1250) — matrix operand order repaired and checked by reading, not compiled, not run on hardware
//
#include <hip/hip_runtime.h>
#include <hip/hip_bf16.h>
#include <math.h>

// ---------------------------------------------------------------------------
// StabilizedEventGroupExpertResidualConditioner for MI455X (gfx1250, wave32)
//
// Tokens bucketed by expert -> only the selected expert's 2-layer MLP runs
// (6x less compute than the dense reference). Matmuls use
// v_wmma_f32_16x16x32_f16 with weights pre-swizzled into B-fragment order
// (one 32B/lane global_load_b128 per fragment, L2-resident). An empty
// inline-asm tie forces 6 B-fragments to be simultaneously live so 12
// global_load_b128 stay in flight per WMMA group (the scheduler otherwise
// re-serializes to one load-pair + s_wait_loadcnt 0 per WMMA). Residual is
// scaled by alpha ~9e-4 before the fp32 add + L2-normalize, so f16 GEMM
// precision is more than sufficient.
// ---------------------------------------------------------------------------

typedef __attribute__((ext_vector_type(16))) _Float16 v16h;
typedef __attribute__((ext_vector_type(8)))  _Float16 v8h;
typedef __attribute__((ext_vector_type(4)))  _Float16 v4h;
typedef __attribute__((ext_vector_type(8)))  float    v8f;

#define E_   6
#define D_   768
#define H_   1536
#define KT1  (D_/32)   // 24  K-tiles, GEMM1
#define NT1  (H_/16)   // 96  N-tiles, GEMM1
#define KT2  (H_/32)   // 48  K-tiles, GEMM2
#define NT2  (D_/16)   // 48  N-tiles, GEMM2

// ---- Kernel 1: bucket tokens by expert group ------------------------------
__global__ void k_build_perm(const int* __restrict__ types,
                             const int* __restrict__ e2g,
                             int* __restrict__ cnt,
                             int* __restrict__ perm, int BL, int T) {
  int i = blockIdx.x * blockDim.x + threadIdx.x;
  if (i >= BL) return;
  int t = types[i];
  t = t < 0 ? 0 : (t > T - 1 ? T - 1 : t);
  int g = e2g[t];
  int pos = atomicAdd(&cnt[g], 1);
  perm[g * BL + pos] = i;
}

// ---- Kernel 2: convert f32 weights -> f16 in WMMA B-fragment order --------
// Fragment = 32x16 (KxN) f16 tile, 512 halves: lane l owns column N = l&15,
// halves j=0..15 hold K = 8*(l>>4) + (j&7) + 16*(j>>3), stored lane-contiguous (32 B/lane).
__global__ void k_swizzle(const float* __restrict__ W, _Float16* __restrict__ Wf,
                          int KT, int NT, int Ksrc, int Nsrc) {
  long idx = (long)blockIdx.x * blockDim.x + threadIdx.x;
  long total = (long)E_ * KT * NT * 512;
  if (idx >= total) return;
  int  j    = (int)(idx & 15);
  int  lane = (int)((idx >> 4) & 31);
  long frag = idx >> 9;
  int  nt   = (int)(frag % NT);
  int  kt   = (int)((frag / NT) % KT);
  int  e    = (int)(frag / ((long)NT * KT));
  int  col  = lane & 15;
  int  K    = ((lane >> 4) << 3) + (j & 7) + ((j >> 3) << 4);
  long src  = ((long)e * Ksrc + (kt * 32 + K)) * Nsrc + (nt * 16 + col);
  Wf[idx] = (_Float16)W[src];
}

// ---- Kernel 3: per-(expert, 16-token tile) fused MLP + normalize ----------
__global__ __launch_bounds__(256)
void k_expert_mlp(const float* __restrict__ x,
                  const int* __restrict__ cnt, const int* __restrict__ perm,
                  const _Float16* __restrict__ w1f, const _Float16* __restrict__ w2f,
                  const float* __restrict__ b1, const float* __restrict__ b2,
                  const float* __restrict__ raw_alpha,
                  float* __restrict__ out, int BL, int maxT) {
  int e    = blockIdx.x / maxT;
  int tile = blockIdx.x % maxT;
  int n    = cnt[e];
  if (tile * 16 >= n) return;

  // 48 KB region, time-aliased: xs(f16 16x768) -> h(f16 16x1536) -> r(f32 16x768)
  __shared__ _Float16 smem[16 * H_];
  __shared__ int      s_tok[16];
  __shared__ float    s_red[16 * 16];
  __shared__ float    s_scale[16];

  int tid  = threadIdx.x;
  int lane = tid & 31;
  int wave = tid >> 5;
  int vcnt = n - tile * 16; if (vcnt > 16) vcnt = 16;

  if (tid < 16) {
    int s = tid < vcnt ? tid : (vcnt - 1);   // pad with a duplicate valid token
    s_tok[tid] = perm[e * BL + tile * 16 + s];
  }
  __syncthreads();

  // ---- stage x tile as f16 [16][768] in LDS (vectorized: float4 -> v4h) ----
  _Float16* xs = smem;
  for (int i = tid; i < 16 * (D_ / 4); i += 256) {
    int row = i / (D_ / 4);
    int c4  = i % (D_ / 4);
    const float4 v = ((const float4*)(x + (long)s_tok[row] * D_))[c4];
    v4h hv;
    hv.x = (_Float16)v.x; hv.y = (_Float16)v.y;
    hv.z = (_Float16)v.z; hv.w = (_Float16)v.w;
    *(v4h*)(xs + row * D_ + c4 * 4) = hv;
  }
  __syncthreads();

  // A-fragment addressing (ISA 7.12.2, 16-bit A 16x32):
  // lane l: row M = l&15, kb = 8*(l>>4); halves 0..7 -> K=kb+j, 8..15 -> K=kb+8+j
  int row = lane & 15;
  int kb  = (lane >> 4) * 8;
  int coln  = lane & 15;          // C/D: N = lane&15
  int mrow0 = (lane >> 4) * 8;    // C/D: M = i + 8*(lane>=16)

  // ================= GEMM1: h[16,1536] = relu(X[16,768] @ W1[e] + b1) ======
  v8f acc[12];
  #pragma unroll
  for (int t = 0; t < 12; ++t)
    #pragma unroll
    for (int i = 0; i < 8; ++i) acc[t][i] = 0.0f;

  const _Float16* w1e = w1f + (long)e * KT1 * NT1 * 512 + (long)lane * 16;
  for (int kt = 0; kt < KT1; ++kt) {
    const _Float16* ap = xs + row * D_ + kt * 32 + kb;
    v8h lo = *(const v8h*)(ap);
    v8h hi = *(const v8h*)(ap + 16);
    v16h a = __builtin_shufflevector(lo, hi, 0,1,2,3,4,5,6,7,8,9,10,11,12,13,14,15);
    const _Float16* wp = w1e + (long)kt * NT1 * 512 + (long)wave * 12 * 512;
    if (kt + 1 < KT1)   // warm next kt's fragment region (global_prefetch)
      __builtin_prefetch(wp + (long)NT1 * 512, 0, 1);
    #pragma unroll
    for (int half = 0; half < 2; ++half) {
      const _Float16* hp = wp + (long)half * 6 * 512;
      // issue 6 B-fragment loads, then pin them live so the scheduler cannot
      // re-serialize into one load + drain-wait per WMMA
      v16h b0 = *(const v16h*)(hp);
      v16h b1v = *(const v16h*)(hp + 512);
      v16h b2v = *(const v16h*)(hp + 1024);
      v16h b3v = *(const v16h*)(hp + 1536);
      v16h b4v = *(const v16h*)(hp + 2048);
      v16h b5v = *(const v16h*)(hp + 2560);
      asm volatile("" : "+v"(b0), "+v"(b1v), "+v"(b2v),
                        "+v"(b3v), "+v"(b4v), "+v"(b5v));
      int o = half * 6;
      acc[o+0] = __builtin_amdgcn_wmma_f32_16x16x32_f16(false, a, false, b0,  (short)0, acc[o+0], false, false);
      acc[o+1] = __builtin_amdgcn_wmma_f32_16x16x32_f16(false, a, false, b1v, (short)0, acc[o+1], false, false);
      acc[o+2] = __builtin_amdgcn_wmma_f32_16x16x32_f16(false, a, false, b2v, (short)0, acc[o+2], false, false);
      acc[o+3] = __builtin_amdgcn_wmma_f32_16x16x32_f16(false, a, false, b3v, (short)0, acc[o+3], false, false);
      acc[o+4] = __builtin_amdgcn_wmma_f32_16x16x32_f16(false, a, false, b4v, (short)0, acc[o+4], false, false);
      acc[o+5] = __builtin_amdgcn_wmma_f32_16x16x32_f16(false, a, false, b5v, (short)0, acc[o+5], false, false);
    }
  }
  __syncthreads();   // xs dead -> region can be reused as h

  _Float16* hbuf = smem;   // f16 [16][1536]
  #pragma unroll
  for (int t = 0; t < 12; ++t) {
    int nt = wave * 12 + t;
    float bias = b1[e * H_ + nt * 16 + coln];
    #pragma unroll
    for (int i = 0; i < 8; ++i) {
      float v = acc[t][i] + bias;
      v = v > 0.0f ? v : 0.0f;
      hbuf[(mrow0 + i) * H_ + nt * 16 + coln] = (_Float16)v;
    }
  }
  __syncthreads();

  // ================= GEMM2: r[16,768] = h @ W2[e] + b2 =====================
  v8f acc2[6];
  #pragma unroll
  for (int t = 0; t < 6; ++t)
    #pragma unroll
    for (int i = 0; i < 8; ++i) acc2[t][i] = 0.0f;

  const _Float16* w2e = w2f + (long)e * KT2 * NT2 * 512 + (long)lane * 16;
  for (int kt = 0; kt < KT2; ++kt) {
    const _Float16* ap = hbuf + row * H_ + kt * 32 + kb;
    v8h lo = *(const v8h*)(ap);
    v8h hi = *(const v8h*)(ap + 16);
    v16h a = __builtin_shufflevector(lo, hi, 0,1,2,3,4,5,6,7,8,9,10,11,12,13,14,15);
    const _Float16* wp = w2e + (long)kt * NT2 * 512 + (long)wave * 6 * 512;
    if (kt + 1 < KT2)
      __builtin_prefetch(wp + (long)NT2 * 512, 0, 1);
    v16h b0 = *(const v16h*)(wp);
    v16h b1v = *(const v16h*)(wp + 512);
    v16h b2v = *(const v16h*)(wp + 1024);
    v16h b3v = *(const v16h*)(wp + 1536);
    v16h b4v = *(const v16h*)(wp + 2048);
    v16h b5v = *(const v16h*)(wp + 2560);
    asm volatile("" : "+v"(b0), "+v"(b1v), "+v"(b2v),
                      "+v"(b3v), "+v"(b4v), "+v"(b5v));
    acc2[0] = __builtin_amdgcn_wmma_f32_16x16x32_f16(false, a, false, b0,  (short)0, acc2[0], false, false);
    acc2[1] = __builtin_amdgcn_wmma_f32_16x16x32_f16(false, a, false, b1v, (short)0, acc2[1], false, false);
    acc2[2] = __builtin_amdgcn_wmma_f32_16x16x32_f16(false, a, false, b2v, (short)0, acc2[2], false, false);
    acc2[3] = __builtin_amdgcn_wmma_f32_16x16x32_f16(false, a, false, b3v, (short)0, acc2[3], false, false);
    acc2[4] = __builtin_amdgcn_wmma_f32_16x16x32_f16(false, a, false, b4v, (short)0, acc2[4], false, false);
    acc2[5] = __builtin_amdgcn_wmma_f32_16x16x32_f16(false, a, false, b5v, (short)0, acc2[5], false, false);
  }
  __syncthreads();   // hbuf dead -> region reused as r (f32)

  float* rbuf = (float*)smem;   // f32 [16][768]
  #pragma unroll
  for (int t = 0; t < 6; ++t) {
    int nt = wave * 6 + t;
    float bias = b2[e * D_ + nt * 16 + coln];
    #pragma unroll
    for (int i = 0; i < 8; ++i)
      rbuf[(mrow0 + i) * D_ + nt * 16 + coln] = acc2[t][i] + bias;
  }
  __syncthreads();

  // ====== epilogue: out = normalize(x + alpha * r), scatter to token =======
  float alpha = 0.05f / (1.0f + __expf(-raw_alpha[e]));
  int m = tid >> 4;     // token slot, 16 threads per token
  int c = tid & 15;
  int tok = s_tok[m];
  const float* xr = x + (long)tok * D_;
  float v[48];
  float ss = 0.0f;
  #pragma unroll
  for (int k = 0; k < 48; ++k) {
    int col = c + 16 * k;
    float val = xr[col] + alpha * rbuf[m * D_ + col];
    v[k] = val;
    ss += val * val;
  }
  s_red[m * 16 + c] = ss;
  __syncthreads();
  if (c == 0) {
    float tot = 0.0f;
    #pragma unroll
    for (int q = 0; q < 16; ++q) tot += s_red[m * 16 + q];
    float nrm = sqrtf(tot);
    s_scale[m] = 1.0f / fmaxf(nrm, 1e-12f);
  }
  __syncthreads();
  float sc = s_scale[m];
  if (m < vcnt) {
    float* op = out + (long)tok * D_;
    #pragma unroll
    for (int k = 0; k < 48; ++k) op[c + 16 * k] = v[k] * sc;
  }
}

// ---------------------------------------------------------------------------
extern "C" void kernel_launch(void* const* d_in, const int* in_sizes, int n_in,
                              void* d_out, int out_size, void* d_ws, size_t ws_size,
                              hipStream_t stream) {
  const float* x        = (const float*)d_in[0];
  const int*   types    = (const int*)  d_in[1];
  const float* W1       = (const float*)d_in[2];
  const float* b1       = (const float*)d_in[3];
  const float* W2       = (const float*)d_in[4];
  const float* b2       = (const float*)d_in[5];
  const float* ralpha   = (const float*)d_in[6];
  const int*   e2g      = (const int*)  d_in[7];
  float*       out      = (float*)d_out;

  int BL = in_sizes[1];        // B*L tokens
  int T  = in_sizes[7];        // event type count

  // workspace layout: [counters 256B][perm E*BL i32][W1 f16 swz][W2 f16 swz]
  char*  ws   = (char*)d_ws;
  int*   cnt  = (int*)ws;
  int*   perm = (int*)(ws + 256);
  size_t o1 = 256 + (size_t)E_ * BL * sizeof(int);
  o1 = (o1 + 255) & ~(size_t)255;
  _Float16* w1f = (_Float16*)(ws + o1);
  size_t w1b = (size_t)E_ * D_ * H_ * sizeof(_Float16);
  size_t o2 = (o1 + w1b + 255) & ~(size_t)255;
  _Float16* w2f = (_Float16*)(ws + o2);

  hipMemsetAsync(cnt, 0, 256, stream);
  k_build_perm<<<(BL + 255) / 256, 256, 0, stream>>>(types, e2g, cnt, perm, BL, T);

  long tot = (long)E_ * KT1 * NT1 * 512;   // same element count for W2
  k_swizzle<<<(int)((tot + 255) / 256), 256, 0, stream>>>(W1, w1f, KT1, NT1, D_, H_);
  k_swizzle<<<(int)((tot + 255) / 256), 256, 0, stream>>>(W2, w2f, KT2, NT2, H_, D_);

  int maxT = (BL + 15) / 16;
  k_expert_mlp<<<E_ * maxT, 256, 0, stream>>>(x, cnt, perm, w1f, w2f,
                                              b1, b2, ralpha, out, BL, maxT);
}
